// ProofGNN_NextTactic_60601988546936
// MI455X (gfx1250) — matrix operand order repaired and checked by reading, not compiled, not run on hardware
//
#include <hip/hip_runtime.h>

typedef __attribute__((ext_vector_type(16))) _Float16 v16h;
typedef __attribute__((ext_vector_type(8)))  _Float16 v8h;
typedef __attribute__((ext_vector_type(8)))  float    v8f;

#define NN   20000
#define EE   320000
#define GG   256
#define NTAC 1000
#define IND  96
#define HD   512

#define NTILES (NN / 16)            // 1250 m-tiles
#define MTPB   8                    // m-tiles per block (4 waves x 2)
#define MBLK   ((NTILES + MTPB - 1) / MTPB)   // 157
#define MPAD   (MBLK * MTPB * 16)   // 20096 padded rows for activation buffers

// ---------------- utility kernels ----------------

__global__ void k_zero(float* __restrict__ p, int n) {
    int i = blockIdx.x * blockDim.x + threadIdx.x;
    if (i < n) p[i] = 0.0f;
}

__global__ void k_f32_to_f16(const float* __restrict__ s, _Float16* __restrict__ d, int n) {
    int i = blockIdx.x * blockDim.x + threadIdx.x;
    if (i < n) d[i] = (_Float16)s[i];
}

// embedding gather + concat -> x16 [N, 96] f16
__global__ void k_embed(const int* __restrict__ ntype, const int* __restrict__ ntac,
                        const float* __restrict__ type_emb, const float* __restrict__ tac_emb,
                        _Float16* __restrict__ x16) {
    int i = blockIdx.x * blockDim.x + threadIdx.x;
    if (i >= NN * IND) return;
    int n = i / IND, d = i % IND;
    float v;
    if (d < 32) {
        v = type_emb[ntype[n] * 32 + d];
    } else {
        int s = ntac[n] + 1;
        s = s < 0 ? 0 : (s > NTAC ? NTAC : s);
        v = tac_emb[s * 64 + (d - 32)];
    }
    x16[i] = (_Float16)v;
}

__global__ void k_deg(const int* __restrict__ dst, float* __restrict__ deg) {
    int e = blockIdx.x * blockDim.x + threadIdx.x;
    if (e < EE) atomicAdd(&deg[dst[e]], 1.0f);
}

// per (edge, dim) scatter-add: agg[dst] += x[src]
__global__ void k_scatter(const int* __restrict__ src, const int* __restrict__ dst,
                          const _Float16* __restrict__ x, float* __restrict__ agg,
                          int D, long total) {
    long i = (long)blockIdx.x * blockDim.x + threadIdx.x;
    if (i >= total) return;
    int e = (int)(i / D), d = (int)(i % D);
    atomicAdd(&agg[(long)dst[e] * D + d], (float)x[(long)src[e] * D + d]);
}

__global__ void k_mean_cvt(const float* __restrict__ agg, const float* __restrict__ deg,
                           _Float16* __restrict__ out, int D, int total) {
    int i = blockIdx.x * blockDim.x + threadIdx.x;
    if (i >= total) return;
    int n = i / D;
    float c = deg[n];
    c = c < 1.0f ? 1.0f : c;
    out[i] = (_Float16)(agg[i] / c);
}

// ------------- fused dual-GEMM WMMA kernel (128M x 64N per block) -------------
// C[M, 512] = A1[M,K] @ B1[512,K]^T + A2[M,K] @ B2[512,K]^T + bias
// Block = 4 waves sharing a 64-column weight slice of B1/B2, staged in LDS in
// K-chunks of KC<=256 (64 KB max -> 5 blocks/WGP). Each wave owns a 32M x 64N
// tile (2 m-subtiles x 4 n-subtiles): every B fragment from LDS feeds TWO
// WMMAs, each A fragment feeds FOUR. 16 WMMAs per K-step per wave.
// A-side buffers are padded to MPAD rows so tail waves never branch around the
// chunk barriers; only the epilogue is guarded (wave-uniformly).
// mode 0: relu -> out16 f16                                  (layer 1)
// mode 1: atomicAdd into gsum[batch[m]*512 + n]              (layer 2 + pool)
__global__ __launch_bounds__(128)
void k_gemm_dual(const _Float16* __restrict__ A1, const _Float16* __restrict__ B1,
                 const _Float16* __restrict__ A2, const _Float16* __restrict__ B2,
                 const float* __restrict__ bias, int K, int mode,
                 _Float16* __restrict__ out16, float* __restrict__ gsum,
                 const int* __restrict__ batch) {
    extern __shared__ _Float16 bsm[];           // [2][64][KC]

    const int NT = HD / 64;                     // 8 column super-tiles
    const int KC = K <= 256 ? K : 256;          // K-chunk staged in LDS
    int wave = threadIdx.x >> 5;
    int lane = threadIdx.x & 31;
    int nt   = blockIdx.x % NT;
    int mblk = blockIdx.x / NT;
    int n0 = nt * 64;
    int mt0 = mblk * MTPB + wave * 2;           // wave-uniform pair of m-tiles
    int m0 = mt0 * 16;
    int lo = lane & 15, hi = lane >> 4;

    v8f acc[2][4];
#pragma unroll
    for (int t = 0; t < 2; t++)
#pragma unroll
        for (int j = 0; j < 4; j++) acc[t][j] = (v8f){};

    const _Float16* a1r0 = A1 + (long)(m0 + lo) * K;
    const _Float16* a1r1 = A1 + (long)(m0 + 16 + lo) * K;
    const _Float16* a2r0 = A2 + (long)(m0 + lo) * K;
    const _Float16* a2r1 = A2 + (long)(m0 + 16 + lo) * K;
    const _Float16* b1lds = bsm + (long)lo * KC;            // sub-tile j adds j*16*KC
    const _Float16* b2lds = bsm + (long)(64 + lo) * KC;

    for (int kb = 0; kb < K; kb += KC) {
        // ---- cooperative stage of B1/B2 64-col x KC slice into LDS ----
        {
            int kc8 = KC / 8;
            int chunksPerMat = 64 * kc8;        // 8-half (16B) chunks
            for (int id = threadIdx.x; id < 2 * chunksPerMat; id += 128) {
                int mat = id / chunksPerMat;
                int rem = id % chunksPerMat;
                int r = rem / kc8, cc = rem % kc8;
                const _Float16* srcp = (mat ? B2 : B1) + (long)(n0 + r) * K + kb + cc * 8;
                *(v8h*)(bsm + (long)mat * 64 * KC + (long)r * KC + cc * 8) = *(const v8h*)srcp;
            }
        }
        __syncthreads();

        for (int k0 = 0; k0 < KC; k0 += 32) {
            // A fragments (contiguous k = 16*hi + s, matching B) for both m-subtiles
            v8h lo0, hi0;
            v16h a1f0, a1f1, a2f0, a2f1;
            lo0 = *(const v8h*)(a1r0 + kb + k0 + hi * 16);
            hi0 = *(const v8h*)(a1r0 + kb + k0 + 8 + hi * 16);
#pragma unroll
            for (int i = 0; i < 8; i++) { a1f0[i] = lo0[i]; a1f0[8 + i] = hi0[i]; }
            lo0 = *(const v8h*)(a1r1 + kb + k0 + hi * 16);
            hi0 = *(const v8h*)(a1r1 + kb + k0 + 8 + hi * 16);
#pragma unroll
            for (int i = 0; i < 8; i++) { a1f1[i] = lo0[i]; a1f1[8 + i] = hi0[i]; }
            lo0 = *(const v8h*)(a2r0 + kb + k0 + hi * 16);
            hi0 = *(const v8h*)(a2r0 + kb + k0 + 8 + hi * 16);
#pragma unroll
            for (int i = 0; i < 8; i++) { a2f0[i] = lo0[i]; a2f0[8 + i] = hi0[i]; }
            lo0 = *(const v8h*)(a2r1 + kb + k0 + hi * 16);
            hi0 = *(const v8h*)(a2r1 + kb + k0 + 8 + hi * 16);
#pragma unroll
            for (int i = 0; i < 8; i++) { a2f1[i] = lo0[i]; a2f1[8 + i] = hi0[i]; }

#pragma unroll
            for (int j = 0; j < 4; j++) {
                // B fragment feeds both m-subtiles
                v16h b1 = *(const v16h*)(b1lds + (long)j * 16 * KC + k0 + hi * 16);
                acc[0][j] = __builtin_amdgcn_wmma_f32_16x16x32_f16(false, a1f0, false, b1,
                                                                   (short)0, acc[0][j], false, false);
                acc[1][j] = __builtin_amdgcn_wmma_f32_16x16x32_f16(false, a1f1, false, b1,
                                                                   (short)0, acc[1][j], false, false);
                v16h b2 = *(const v16h*)(b2lds + (long)j * 16 * KC + k0 + hi * 16);
                acc[0][j] = __builtin_amdgcn_wmma_f32_16x16x32_f16(false, a2f0, false, b2,
                                                                   (short)0, acc[0][j], false, false);
                acc[1][j] = __builtin_amdgcn_wmma_f32_16x16x32_f16(false, a2f1, false, b2,
                                                                   (short)0, acc[1][j], false, false);
            }
        }
        __syncthreads();                        // protect LDS before next chunk
    }

#pragma unroll
    for (int t = 0; t < 2; t++) {
        if (mt0 + t >= NTILES) continue;        // wave-uniform tail guard
        int m0t = (mt0 + t) * 16;
        if (mode == 0) {
#pragma unroll
            for (int j = 0; j < 4; j++) {
                int n = n0 + j * 16 + lo;
                float bv = bias[n];
#pragma unroll
                for (int v = 0; v < 8; v++) {
                    int m = m0t + v + 8 * hi;   // C layout: VGPR v -> row v + 8*hi, col lo
                    float val = acc[t][j][v] + bv;
                    val = val > 0.0f ? val : 0.0f;
                    out16[(long)m * HD + n] = (_Float16)val;
                }
            }
        } else {
#pragma unroll
            for (int j = 0; j < 4; j++) {
                int n = n0 + j * 16 + lo;
                float bv = bias[n];
#pragma unroll
                for (int v = 0; v < 8; v++) {
                    int m = m0t + v + 8 * hi;
                    atomicAdd(&gsum[(long)batch[m] * HD + n], acc[t][j][v] + bv);
                }
            }
        }
    }
}

__global__ void k_gcnt(const int* __restrict__ batch, float* __restrict__ gcnt) {
    int n = blockIdx.x * blockDim.x + threadIdx.x;
    if (n < NN) atomicAdd(&gcnt[batch[n]], 1.0f);
}

// h = relu((gsum/gcnt) @ Wc1^T + bc1)
__global__ void k_cls1(const float* __restrict__ gsum, const float* __restrict__ gcnt,
                       const float* __restrict__ Wc1, const float* __restrict__ bc1,
                       float* __restrict__ hc) {
    int i = blockIdx.x * blockDim.x + threadIdx.x;
    if (i >= GG * HD) return;
    int g = i / HD, j = i % HD;
    float cnt = gcnt[g];
    float sc = 1.0f / (cnt < 1.0f ? 1.0f : cnt);
    float acc = 0.0f;
    const float* gr = gsum + (long)g * HD;
    const float* w  = Wc1 + (long)j * HD;
    for (int k = 0; k < HD; k++) acc += gr[k] * w[k];
    acc = acc * sc + bc1[j];
    hc[i] = acc > 0.0f ? acc : 0.0f;
}

// logits = hc @ Wc2^T + bc2
__global__ void k_cls2(const float* __restrict__ hc, const float* __restrict__ Wc2,
                       const float* __restrict__ bc2, float* __restrict__ out) {
    int i = blockIdx.x * blockDim.x + threadIdx.x;
    if (i >= GG * NTAC) return;
    int g = i / NTAC, t = i % NTAC;
    float acc = bc2[t];
    const float* h = hc  + (long)g * HD;
    const float* w = Wc2 + (long)t * HD;
    for (int k = 0; k < HD; k++) acc += h[k] * w[k];
    out[i] = acc;
}

// ---------------- host launcher ----------------

static inline size_t alignup(size_t x) { return (x + 255) & ~(size_t)255; }

extern "C" void kernel_launch(void* const* d_in, const int* in_sizes, int n_in,
                              void* d_out, int out_size, void* d_ws, size_t ws_size,
                              hipStream_t stream) {
    (void)in_sizes; (void)n_in; (void)out_size; (void)ws_size;

    const int*   node_type  = (const int*)d_in[0];
    const int*   node_tac   = (const int*)d_in[1];
    const int*   edge_index = (const int*)d_in[2];   // [2, E] row-major
    const int*   batch      = (const int*)d_in[3];
    const float* type_emb   = (const float*)d_in[4];
    const float* tactic_emb = (const float*)d_in[5];
    const float* W1l = (const float*)d_in[6];
    const float* b1l = (const float*)d_in[7];
    const float* W1r = (const float*)d_in[8];
    const float* W2l = (const float*)d_in[9];
    const float* b2l = (const float*)d_in[10];
    const float* W2r = (const float*)d_in[11];
    const float* Wc1 = (const float*)d_in[12];
    const float* bc1 = (const float*)d_in[13];
    const float* Wc2 = (const float*)d_in[14];
    const float* bc2 = (const float*)d_in[15];
    float* out = (float*)d_out;

    const int* e_src = edge_index;
    const int* e_dst = edge_index + EE;

    // workspace carve-up (A-side activation buffers padded to MPAD rows)
    char* ws = (char*)d_ws;
    size_t off = 0;
    auto take = [&](size_t bytes) { char* p = ws + off; off += alignup(bytes); return p; };

    _Float16* x16    = (_Float16*)take((size_t)MPAD * IND * 2);
    float*    agg1f  = (float*)   take((size_t)NN * IND * 4);
    _Float16* agg1h  = (_Float16*)take((size_t)MPAD * IND * 2);
    float*    deg    = (float*)   take((size_t)NN * 4);
    _Float16* h1     = (_Float16*)take((size_t)MPAD * HD * 2);
    float*    agg2f  = (float*)   take((size_t)NN * HD * 4);
    _Float16* agg2h  = (_Float16*)take((size_t)MPAD * HD * 2);
    float*    gsum   = (float*)   take((size_t)GG * HD * 4);
    float*    gcnt   = (float*)   take((size_t)GG * 4);
    float*    hc     = (float*)   take((size_t)GG * HD * 4);
    _Float16* w1l16  = (_Float16*)take((size_t)HD * IND * 2);
    _Float16* w1r16  = (_Float16*)take((size_t)HD * IND * 2);
    _Float16* w2l16  = (_Float16*)take((size_t)HD * HD * 2);
    _Float16* w2r16  = (_Float16*)take((size_t)HD * HD * 2);

    const int T = 256;
    auto blks = [](long n, int t) { return (unsigned)((n + t - 1) / t); };

    // zero accumulators
    k_zero<<<blks(NN * IND, T), T, 0, stream>>>(agg1f, NN * IND);
    k_zero<<<blks((long)NN * HD, T), T, 0, stream>>>(agg2f, NN * HD);
    k_zero<<<blks(NN, T), T, 0, stream>>>(deg, NN);
    k_zero<<<blks(GG * HD, T), T, 0, stream>>>(gsum, GG * HD);
    k_zero<<<blks(GG, T), T, 0, stream>>>(gcnt, GG);

    // weight conversion f32 -> f16
    k_f32_to_f16<<<blks(HD * IND, T), T, 0, stream>>>(W1l, w1l16, HD * IND);
    k_f32_to_f16<<<blks(HD * IND, T), T, 0, stream>>>(W1r, w1r16, HD * IND);
    k_f32_to_f16<<<blks(HD * HD, T), T, 0, stream>>>(W2l, w2l16, HD * HD);
    k_f32_to_f16<<<blks(HD * HD, T), T, 0, stream>>>(W2r, w2r16, HD * HD);

    // embeddings
    k_embed<<<blks(NN * IND, T), T, 0, stream>>>(node_type, node_tac, type_emb, tactic_emb, x16);

    // degree
    k_deg<<<blks(EE, T), T, 0, stream>>>(e_dst, deg);

    // layer 1: aggregate neighbors (96-d)
    k_scatter<<<blks((long)EE * IND, T), T, 0, stream>>>(e_src, e_dst, x16, agg1f, IND, (long)EE * IND);
    k_mean_cvt<<<blks(NN * IND, T), T, 0, stream>>>(agg1f, deg, agg1h, IND, NN * IND);

    // GEMM grid: 157 m-blocks (8 m-tiles each) x 8 column super-tiles
    const unsigned gemm_blocks = MBLK * (HD / 64);       // 1256

    // layer 1 GEMM: h1 = relu(x@W1r^T + agg@W1l^T + b1l)   [WMMA, B in LDS]
    k_gemm_dual<<<gemm_blocks, 128, (size_t)256 * IND, stream>>>(
        x16, w1r16, agg1h, w1l16, b1l, IND, 0, h1, nullptr, nullptr);

    // layer 2: aggregate neighbors (512-d)
    k_scatter<<<blks((long)EE * HD, T), T, 0, stream>>>(e_src, e_dst, h1, agg2f, HD, (long)EE * HD);
    k_mean_cvt<<<blks((long)NN * HD, T), T, 0, stream>>>(agg2f, deg, agg2h, HD, NN * HD);

    // layer 2 GEMM + fused graph-sum pooling   [WMMA, B in LDS, KC=256 chunks]
    k_gemm_dual<<<gemm_blocks, 128, (size_t)256 * 256, stream>>>(
        h1, w2r16, agg2h, w2l16, b2l, HD, 1, nullptr, gsum, batch);

    // pooling denominators + classifier
    k_gcnt<<<blks(NN, T), T, 0, stream>>>(batch, gcnt);
    k_cls1<<<blks(GG * HD, T), T, 0, stream>>>(gsum, gcnt, Wc1, bc1, hc);
    k_cls2<<<blks(GG * NTAC, T), T, 0, stream>>>(hc, Wc2, bc2, out);
}
